// FFAS_78288663872237
// MI455X (gfx1250) — matrix operand order repaired and checked by reading, not compiled, not run on hardware
//
#include <hip/hip_runtime.h>

typedef __attribute__((ext_vector_type(16))) __bf16 v16bf;
typedef __attribute__((ext_vector_type(8)))  float  v8f;
typedef unsigned int u32x4 __attribute__((ext_vector_type(4)));
typedef int          i32x4 __attribute__((ext_vector_type(4)));
typedef int          i32x8 __attribute__((ext_vector_type(8)));

union F16x16 {
    v16bf v;
    uint4 q[2];
    unsigned short h[16];
};

// --- TDM availability / arity detection ---------------------------------
#if defined(__has_builtin)
#if __has_builtin(__builtin_amdgcn_tensor_load_to_lds) && \
    __has_builtin(__builtin_amdgcn_s_wait_tensorcnt)
#define HAVE_TDM 1
#endif
#endif
#ifndef HAVE_TDM
#define HAVE_TDM 0
#endif
#if defined(__clang_major__) && (__clang_major__ >= 23)
#define CDNA5_TDM_6ARG 1
#else
#define CDNA5_TDM_6ARG 0
#endif

__device__ __forceinline__ unsigned short f2bf(float f) {
    unsigned int u = __float_as_uint(f);
    unsigned int r = (u + 0x7FFFu + ((u >> 16) & 1u)) >> 16;
    return (unsigned short)r;
}

__device__ __forceinline__ v8f vzero8() {
    v8f z = {0.f, 0.f, 0.f, 0.f, 0.f, 0.f, 0.f, 0.f};
    return z;
}

__device__ __forceinline__ unsigned ldsOff(const void* p) {
    // generic pointers to LDS carry the byte offset in the low 32 bits
    return (unsigned)(uintptr_t)p;
}

#if HAVE_TDM
// Issue one TDM 2-D tile load: global (row-major, strideElems) -> LDS.
// tileD0/tensorD0 along contiguous dim, tileD1/tensorD1 rows.
// padI/padA: D# pad_interval / pad_amount codes (LDS destination padding).
__device__ __forceinline__ void tdm_load(unsigned ldsAddr, const unsigned short* gptr,
                                         int tensorD0, int tensorD1, long long strideElems,
                                         int tileD0, int tileD1, int padI, int padA) {
    unsigned long long ga = (unsigned long long)(uintptr_t)gptr;
    u32x4 g0 = {0u, 0u, 0u, 0u};
    g0[0] = 1u;                                              // count=1, user mode
    g0[1] = ldsAddr;                                         // lds_addr
    g0[2] = (unsigned)ga;                                    // global_addr[31:0]
    g0[3] = (unsigned)((ga >> 32) & 0x01ffffffull) | (2u << 30);  // addr[56:32] | type=2
    i32x8 g1 = {0, 0, 0, 0, 0, 0, 0, 0};
    g1[0] = (1 << 16) | (1 << 20) | (padI << 22) | (padA << 25);  // data_size=2B, pad_enable
    g1[1] = (tensorD0 & 0xffff) << 16;                       // tensor_dim0[15:0]
    g1[2] = ((tensorD0 >> 16) & 0xffff) | ((tensorD1 & 0xffff) << 16);
    g1[3] = ((tensorD1 >> 16) & 0xffff) | ((tileD0 & 0xffff) << 16);
    g1[4] = (tileD1 & 0xffff);                               // tile_dim1, tile_dim2=0
    g1[5] = (int)(strideElems & 0xffffffffll);               // tensor_dim0_stride lo
    g1[6] = (int)((strideElems >> 32) & 0xffffll);           // stride hi, dim1_stride=0
    i32x4 z4 = {0, 0, 0, 0};
#if CDNA5_TDM_6ARG
    i32x8 z8 = {0, 0, 0, 0, 0, 0, 0, 0};
    __builtin_amdgcn_tensor_load_to_lds(g0, g1, z4, z4, z8, 0);
#else
    __builtin_amdgcn_tensor_load_to_lds(g0, g1, z4, z4, 0);
#endif
}
#endif

// ---------------------------------------------------------------------------
// Transpose + convert: src[N][K] fp32 -> dst[K][N] bf16 (32x32 LDS tiles)
// ---------------------------------------------------------------------------
__global__ __launch_bounds__(256)
void transpose_f32_bf16(const float* __restrict__ src, unsigned short* __restrict__ dst,
                        int N, int K) {
    __shared__ float tile[32][33];
    const int t = threadIdx.x;
    const int n0 = blockIdx.x * 32;
    const int k0 = blockIdx.y * 32;
    const int r = t >> 3, c = (t & 7) * 4;
    const float* sp = src + (size_t)(n0 + r) * K + k0 + c;
#pragma unroll
    for (int j = 0; j < 4; ++j) tile[r][c + j] = sp[j];
    __syncthreads();
    unsigned short* dp = dst + (size_t)(k0 + r) * N + n0 + c;
#pragma unroll
    for (int j = 0; j < 4; ++j) dp[j] = f2bf(tile[c + j][r]);
}

// ---------------------------------------------------------------------------
// LayerNorm over E=768, output bf16
// ---------------------------------------------------------------------------
__global__ __launch_bounds__(256)
void ln_bf16(const float* __restrict__ x, const float* __restrict__ w,
             const float* __restrict__ b, unsigned short* __restrict__ y) {
    const int row = blockIdx.x;
    const int t = threadIdx.x;
    const float* xp = x + (size_t)row * 768;
    float v0 = xp[t], v1 = xp[t + 256], v2 = xp[t + 512];
    float s  = v0 + v1 + v2;
    float s2 = v0 * v0 + v1 * v1 + v2 * v2;
#pragma unroll
    for (int m = 16; m >= 1; m >>= 1) {
        s  += __shfl_xor(s, m);
        s2 += __shfl_xor(s2, m);
    }
    __shared__ float red[2][8];
    if ((t & 31) == 0) { red[0][t >> 5] = s; red[1][t >> 5] = s2; }
    __syncthreads();
    float ts = 0.f, ts2 = 0.f;
#pragma unroll
    for (int i = 0; i < 8; ++i) { ts += red[0][i]; ts2 += red[1][i]; }
    const float inv768 = 1.0f / 768.0f;
    float mean = ts * inv768;
    float var  = ts2 * inv768 - mean * mean;
    float rstd = rsqrtf(var + 1e-5f);
    unsigned short* yp = y + (size_t)row * 768;
    yp[t]       = f2bf((v0 - mean) * rstd * w[t]       + b[t]);
    yp[t + 256] = f2bf((v1 - mean) * rstd * w[t + 256] + b[t + 256]);
    yp[t + 512] = f2bf((v2 - mean) * rstd * w[t + 512] + b[t + 512]);
}

// ---------------------------------------------------------------------------
// Tiled WMMA GEMM: C[M,N] = A_bf16[M,K] * Wt_bf16[K,N]  (+ epilogue)
// Tiles staged into LDS by the Tensor Data Mover (wave 0 issues, TENSORcnt +
// block barrier synchronize), double-buffered. 8 waves, each 32x32 of 64x128.
// MODE 0: qkv scatter (q scaled, k transposed, v plain), bias = in_proj_b
// MODE 1: outF = acc + bias[n] + resid[row,n]          (fp32)
// MODE 2: outB = quickgelu(acc + bias[n])              (bf16)
// ---------------------------------------------------------------------------
template <int MODE>
__global__ __launch_bounds__(256)
void gemm_wmma(const unsigned short* __restrict__ A,   // [M,K] bf16
               const unsigned short* __restrict__ Wt,  // [K,N] bf16 (pre-transposed)
               const float* __restrict__ bias,
               const float* __restrict__ resid,
               float* __restrict__ outF,
               unsigned short* __restrict__ outB,
               unsigned short* __restrict__ qOut,
               unsigned short* __restrict__ kOut,
               unsigned short* __restrict__ vOut,
               int M, int N, int K) {
    __shared__ __align__(16) unsigned short sA[2][64][40];    // 64x32 + 16B row pad
    __shared__ __align__(16) unsigned short sB[2][32][136];   // 32x128 + 16B row pad

    const int t = threadIdx.x;
    const int lane = t & 31;
    const int wave = t >> 5;
    const int waveM = wave & 1;   // 2 waves over M (64 rows)
    const int waveN = wave >> 1;  // 4 waves over N (128 cols)
    const int hi = lane >> 4, lo = lane & 15;
    const int bM = blockIdx.x * 64;
    const int bN = blockIdx.y * 128;
    const int rowsA = M - bM;

    auto issue = [&](int buf, int k0) {
#if HAVE_TDM
        if (t < 32) {
            // A tile: 64 rows x 32 cols, zero-fill rows beyond M via tensor_dim1
            tdm_load(ldsOff(&sA[buf][0][0]), A + (size_t)bM * K + k0,
                     32, rowsA, (long long)K, 32, 64, /*pad 16DW*/3, /*4DW*/3);
            // B tile: 32 rows x 128 cols from Wt
            tdm_load(ldsOff(&sB[buf][0][0]), Wt + (size_t)k0 * N + bN,
                     128, 32, (long long)N, 128, 32, /*pad 64DW*/5, /*4DW*/3);
        }
#else
        {
            int r = t >> 2, c = (t & 3) * 8;
            int row = bM + r;
            uint4 val = {0u, 0u, 0u, 0u};
            if (row < M) val = *reinterpret_cast<const uint4*>(&A[(size_t)row * K + k0 + c]);
            *reinterpret_cast<uint4*>(&sA[buf][r][c]) = val;
        }
        {
            int rr = t >> 3, cc = (t & 7) * 16;
            const unsigned short* wp = &Wt[(size_t)(k0 + rr) * N + bN + cc];
            *reinterpret_cast<uint4*>(&sB[buf][rr][cc])     = *reinterpret_cast<const uint4*>(wp);
            *reinterpret_cast<uint4*>(&sB[buf][rr][cc + 8]) = *reinterpret_cast<const uint4*>(wp + 8);
        }
#endif
    };

    v8f acc[2][2];
    acc[0][0] = vzero8(); acc[0][1] = vzero8();
    acc[1][0] = vzero8(); acc[1][1] = vzero8();

    issue(0, 0);
    const int nk = K >> 5;
    for (int kb = 0; kb < nk; ++kb) {
        const int buf = kb & 1;
#if HAVE_TDM
        if (t < 32) __builtin_amdgcn_s_wait_tensorcnt(0);
#endif
        __syncthreads();
        if (kb + 1 < nk) issue(buf ^ 1, (kb + 1) << 5);
        if (kb + 2 < nk)
            __builtin_prefetch(&Wt[(size_t)(((kb + 2) << 5) + lane) * N + bN + wave * 16], 0, 1);

        F16x16 af[2], bfr[2];
#pragma unroll
        for (int tm = 0; tm < 2; ++tm) {
            const unsigned short* p = &sA[buf][waveM * 32 + tm * 16 + lo][hi * 8];
            af[tm].q[0] = *reinterpret_cast<const uint4*>(p);
            af[tm].q[1] = *reinterpret_cast<const uint4*>(p + 16);
        }
#pragma unroll
        for (int tn = 0; tn < 2; ++tn) {
            const unsigned short* p = &sB[buf][hi * 8][waveN * 32 + tn * 16 + lo];
            for (int s = 0; s < 16; ++s) bfr[tn].h[s] = p[((s & 7) + (s >> 3) * 16) * 136];
        }
#pragma unroll
        for (int tm = 0; tm < 2; ++tm)
#pragma unroll
            for (int tn = 0; tn < 2; ++tn)
                acc[tm][tn] = __builtin_amdgcn_wmma_f32_16x16x32_bf16(
                    false, af[tm].v, false, bfr[tn].v, (short)0, acc[tm][tn], false, false);
    }

    // epilogue (C layout: lane lo -> col, vgpr r + 8*hi -> row)
#pragma unroll
    for (int tm = 0; tm < 2; ++tm) {
#pragma unroll
        for (int tn = 0; tn < 2; ++tn) {
            const int col = bN + waveN * 32 + tn * 16 + lo;
#pragma unroll
            for (int r = 0; r < 8; ++r) {
                const int row = bM + waveM * 32 + tm * 16 + r + hi * 8;
                if (row >= M) continue;
                float v = acc[tm][tn][r] + bias[col];
                if (MODE == 0) {
                    const int s = row >> 3, bb = row & 7;
                    const int nn = col % 768, which = col / 768;
                    const int hh = nn >> 6, dd = nn & 63;
                    if (which == 0)
                        qOut[(((size_t)bb * 12 + hh) * 1025 + s) * 64 + dd] = f2bf(v * 0.125f);
                    else if (which == 1)
                        kOut[(((size_t)bb * 12 + hh) * 64 + dd) * 1056 + s] = f2bf(v);
                    else
                        vOut[(((size_t)bb * 12 + hh) * 1025 + s) * 64 + dd] = f2bf(v);
                } else if (MODE == 1) {
                    outF[(size_t)row * N + col] = v + resid[(size_t)row * N + col];
                } else {
                    float g = v / (1.0f + __expf(-1.702f * v));
                    outB[(size_t)row * N + col] = f2bf(g);
                }
            }
        }
    }
}

// ---------------------------------------------------------------------------
// Flash attention: one wave per 16 q-rows; K-loop over 32 keys with
// multiplicative mask + online softmax; P routed through LDS for the PV WMMA.
// q: [b,h,s,64] (pre-scaled), kT: [b,h,64,1056], v: [b,h,s,64], o: [s,b,768]
// ---------------------------------------------------------------------------
__global__ __launch_bounds__(256)
void flash_attn(const unsigned short* __restrict__ q,
                const unsigned short* __restrict__ kT,
                const unsigned short* __restrict__ v,
                const float* __restrict__ mask,
                unsigned short* __restrict__ o) {
    const int S = 1025, VS = 1056;
    __shared__ __align__(16) unsigned short sP[8][16][40];

    const int bh = blockIdx.x;            // 0..95
    const int bb = bh / 12, hh = bh % 12;
    const int wave = threadIdx.x >> 5, lane = threadIdx.x & 31;
    const int hi = lane >> 4, lo = lane & 15;
    const int q0 = blockIdx.y * 128 + wave * 16;
    if (q0 >= S) return;

    const unsigned short* qb  = q  + (size_t)bh * S * 64;
    const unsigned short* ktb = kT + (size_t)bh * 64 * VS;
    const unsigned short* vb  = v  + (size_t)bh * S * 64;
    const float* mk = mask + (size_t)bb * (S - 1);

    // Q fragments (A layout): lane -> q-row, halves -> d (two 32-wide chunks)
    F16x16 aQ[2];
    {
        int r = q0 + lo;
        if (r > S - 1) r = S - 1;
        const unsigned short* p = qb + (size_t)r * 64 + hi * 8;
        aQ[0].q[0] = *(const uint4*)p;        aQ[0].q[1] = *(const uint4*)(p + 16);
        aQ[1].q[0] = *(const uint4*)(p + 32); aQ[1].q[1] = *(const uint4*)(p + 48);
    }

    float mrow[8], lrow[8];
    v8f oacc[4];
#pragma unroll
    for (int tn = 0; tn < 4; ++tn) oacc[tn] = vzero8();
#pragma unroll
    for (int r = 0; r < 8; ++r) { mrow[r] = -3.0e38f; lrow[r] = 0.0f; }

    const int nkb = (S + 31) >> 5;  // 33
    for (int kblk = 0; kblk < nkb; ++kblk) {
        const int kBase = kblk * 32;
        v8f st[2];
        // --- scores: two 16x16 tiles (keys kBase..+15, +16..+31) ---
#pragma unroll
        for (int th = 0; th < 2; ++th) {
            v8f sc = vzero8();
#pragma unroll
            for (int kf = 0; kf < 2; ++kf) {
                // B operand: lane -> key (lo), elements -> 16 d's (kT rows, table order)
                const unsigned short* p =
                    ktb + (size_t)(kf * 32 + hi * 8) * VS + kBase + th * 16 + lo;
                F16x16 bK;
                for (int s = 0; s < 16; ++s) bK.h[s] = p[(size_t)((s & 7) + (s >> 3) * 16) * VS];
                sc = __builtin_amdgcn_wmma_f32_16x16x32_bf16(
                    false, aQ[kf].v, false, bK.v, (short)0, sc, false, false);
            }
            const int key = kBase + th * 16 + lo;
            const float mv = (key >= 1 && key < S) ? mk[key - 1] : 1.0f;
#pragma unroll
            for (int r = 0; r < 8; ++r) {
                const int rq = q0 + r + hi * 8;
                float xv = sc[r];
                if (key >= S) xv = -3.0e38f;
                else if (rq >= 1 && key >= 1) xv *= mv;
                sc[r] = xv;
            }
            st[th] = sc;
        }
        // --- online softmax over the 32 columns ---
        float pr[2][8];
#pragma unroll
        for (int r = 0; r < 8; ++r) {
            float tmax = fmaxf(st[0][r], st[1][r]);
#pragma unroll
            for (int mm = 8; mm >= 1; mm >>= 1) tmax = fmaxf(tmax, __shfl_xor(tmax, mm));
            const float mn = fmaxf(mrow[r], tmax);
            const float rescale = __expf(mrow[r] - mn);
            mrow[r] = mn;
            float p0 = __expf(st[0][r] - mn);
            float p1 = __expf(st[1][r] - mn);
            float rs = p0 + p1;
#pragma unroll
            for (int mm = 8; mm >= 1; mm >>= 1) rs += __shfl_xor(rs, mm);
            lrow[r] = lrow[r] * rescale + rs;
            pr[0][r] = p0; pr[1][r] = p1;
#pragma unroll
            for (int tn = 0; tn < 4; ++tn) oacc[tn][r] *= rescale;
        }
        // --- P (C layout) -> LDS -> A layout ---
#pragma unroll
        for (int th = 0; th < 2; ++th)
#pragma unroll
            for (int r = 0; r < 8; ++r)
                sP[wave][r + hi * 8][th * 16 + lo] = f2bf(pr[th][r]);
        F16x16 aP;
        {
            const unsigned short* p = &sP[wave][lo][hi * 8];
            aP.q[0] = *(const uint4*)p;
            aP.q[1] = *(const uint4*)(p + 16);
        }
        // --- o += P @ V : B operand lane -> d, elements -> 16 keys ---
#pragma unroll
        for (int tn = 0; tn < 4; ++tn) {
            const unsigned short* p = vb + tn * 16 + lo;
            F16x16 bV;
            for (int s = 0; s < 16; ++s) {
                int key = kBase + (s & 7) + hi * 8 + (s >> 3) * 16;
                if (key > S - 1) key = S - 1;
                bV.h[s] = p[(size_t)key * 64];
            }
            oacc[tn] = __builtin_amdgcn_wmma_f32_16x16x32_bf16(
                false, aP.v, false, bV.v, (short)0, oacc[tn], false, false);
        }
    }
    // --- normalize and store o[s, b, h*64 + d] as bf16 ---
#pragma unroll
    for (int r = 0; r < 8; ++r) {
        const int rq = q0 + r + hi * 8;
        if (rq >= S) continue;
        const float inv = 1.0f / lrow[r];
        unsigned short* op = o + ((size_t)rq * 8 + bb) * 768 + hh * 64;
#pragma unroll
        for (int tn = 0; tn < 4; ++tn)
            op[tn * 16 + lo] = f2bf(oacc[tn][r] * inv);
    }
}

// ---------------------------------------------------------------------------
// host launcher
// ---------------------------------------------------------------------------
extern "C" void kernel_launch(void* const* d_in, const int* in_sizes, int n_in,
                              void* d_out, int out_size, void* d_ws, size_t ws_size,
                              hipStream_t stream) {
    (void)in_sizes; (void)n_in; (void)out_size; (void)ws_size;
    const int S = 1025, E = 768, FF = 3072;
    const int M = S * 8;  // 8200

    const float* x     = (const float*)d_in[0];
    const float* ln1w  = (const float*)d_in[1];
    const float* ln1b  = (const float*)d_in[2];
    const float* inW   = (const float*)d_in[3];
    const float* inB   = (const float*)d_in[4];
    const float* outW  = (const float*)d_in[5];
    const float* outB  = (const float*)d_in[6];
    const float* ln2w  = (const float*)d_in[7];
    const float* ln2b  = (const float*)d_in[8];
    const float* fcW   = (const float*)d_in[9];
    const float* fcB   = (const float*)d_in[10];
    const float* prW   = (const float*)d_in[11];
    const float* prB   = (const float*)d_in[12];
    const float* amask = (const float*)d_in[13];

    char* ws = (char*)d_ws;
    // bf16 pre-transposed weights: Wt[K][N]
    unsigned short* wIn = (unsigned short*)(ws + 0);          // [768,2304]
    unsigned short* wOu = (unsigned short*)(ws + 3538944);    // [768,768]
    unsigned short* wFc = (unsigned short*)(ws + 4718592);    // [768,3072]
    unsigned short* wPr = (unsigned short*)(ws + 9437184);    // [3072,768]
    // activations
    unsigned short* xln = (unsigned short*)(ws + 14155776);   // [8200,768] bf16
    unsigned short* qB  = (unsigned short*)(ws + 26750976);   // [8,12,1025,64] bf16
    unsigned short* kT  = (unsigned short*)(ws + 39346176);   // [8,12,64,1056] bf16
    unsigned short* vB  = (unsigned short*)(ws + 52322304);   // [8,12,1025,64] bf16
    unsigned short* oB  = (unsigned short*)(ws + 64917504);   // [8200,768] bf16
    float*          x2  = (float*)(ws + 77512704);            // [8200,768] f32
    unsigned short* h1  = xln;                                // reuse for ln2 out
    unsigned short* h2  = qB;                                 // reuse q..o region [8200,3072]
    float* out = (float*)d_out;

    // 1. weights -> bf16, transposed to [K][N]
    transpose_f32_bf16<<<dim3(2304 / 32, 768 / 32),  256, 0, stream>>>(inW,  wIn, 2304, 768);
    transpose_f32_bf16<<<dim3(768 / 32,  768 / 32),  256, 0, stream>>>(outW, wOu, 768,  768);
    transpose_f32_bf16<<<dim3(3072 / 32, 768 / 32),  256, 0, stream>>>(fcW,  wFc, 3072, 768);
    transpose_f32_bf16<<<dim3(768 / 32,  3072 / 32), 256, 0, stream>>>(prW,  wPr, 768,  3072);

    // 2. LN1 -> bf16
    ln_bf16<<<M, 256, 0, stream>>>(x, ln1w, ln1b, xln);

    const int gm = (M + 63) / 64;  // 129

    // 3. QKV projection + head scatter
    gemm_wmma<0><<<dim3(gm, (3 * E) / 128), 256, 0, stream>>>(
        xln, wIn, inB, nullptr, nullptr, nullptr, qB, kT, vB, M, 3 * E, E);

    // 4. flash attention
    flash_attn<<<dim3(96, (S + 127) / 128), 256, 0, stream>>>(qB, kT, vB, amask, oB);

    // 5. output projection + residual -> x2 (fp32)
    gemm_wmma<1><<<dim3(gm, E / 128), 256, 0, stream>>>(
        oB, wOu, outB, x, x2, nullptr, nullptr, nullptr, nullptr, M, E, E);

    // 6. LN2 -> bf16
    ln_bf16<<<M, 256, 0, stream>>>(x2, ln2w, ln2b, h1);

    // 7. fc + QuickGELU -> bf16
    gemm_wmma<2><<<dim3(gm, FF / 128), 256, 0, stream>>>(
        h1, wFc, fcB, nullptr, nullptr, h2, nullptr, nullptr, nullptr, M, FF, E);

    // 8. proj + residual -> d_out (fp32)
    gemm_wmma<1><<<dim3(gm, E / 128), 256, 0, stream>>>(
        h2, wPr, prB, x2, out, nullptr, nullptr, nullptr, nullptr, M, E, FF);
}
